// PriorFusion3D_voxel_85444079386628
// MI455X (gfx1250) — matrix operand order repaired and checked by reading, not compiled, not run on hardware
//
#include <hip/hip_runtime.h>
#include <hip/hip_bf16.h>

typedef __attribute__((ext_vector_type(16))) _Float16 v16h;
typedef __attribute__((ext_vector_type(8)))  _Float16 v8h;
typedef __attribute__((ext_vector_type(8)))  float    v8f;

#define VOX_X 256
#define VOX_Y 256
#define VOX_Z 20
#define VOX_TOT (VOX_X * VOX_Y * VOX_Z)   // 1,310,720
#define OUT_H 200
#define OUT_W 200
#define OUT_Z 16
#define SPATIAL (OUT_H * OUT_W * OUT_Z)   // 640,000
#define NCH 80
#define KDIM 160
#define N_PTS 200000
#define NT 32                              // spatial sites per block

// ---------------- kernel 1: init winner-index grid ----------------
__global__ void init_idxgrid(int* __restrict__ grid) {
    int i = blockIdx.x * blockDim.x + threadIdx.x;
    if (i < VOX_TOT) grid[i] = -1;
}

// ---------------- kernel 2: scatter (last write wins == max point idx) ------
__global__ void scatter_points(const int* __restrict__ coords, int* __restrict__ grid) {
    int p = blockIdx.x * blockDim.x + threadIdx.x;
    if (p >= N_PTS) return;
    int x = coords[p * 4 + 1];
    int y = coords[p * 4 + 2];
    int z = coords[p * 4 + 3];
    atomicMax(&grid[(x * VOX_Y + y) * VOX_Z + z], p);
}

// ---------------- kernel 3: pre-swizzle weights into WMMA A fragments ------
// A fragment (16-bit A 16x32, ISA 7.12.2): lane l holds row M=l%16.
// lanes 0-15:  element j<8 -> K=k0+j,    j>=8 -> K=k0+16+(j-8)
// lanes 16-31: element j<8 -> K=k0+8+j,  j>=8 -> K=k0+24+(j-8)
__global__ void prep_weights(const float* __restrict__ w_conv, const float* __restrict__ b_conv,
                             const float* __restrict__ gamma,  const float* __restrict__ beta,
                             const float* __restrict__ mean,   const float* __restrict__ var,
                             _Float16* __restrict__ A_pre, float* __restrict__ invshift) {
    int tid = blockIdx.x * blockDim.x + threadIdx.x;
    if (tid < 800) {                       // 5 m_tiles * 5 k_steps * 32 lanes
        int m_tile = tid / 160;
        int rem    = tid % 160;
        int ks     = rem / 32;
        int lane   = rem % 32;
        int o  = m_tile * 16 + (lane & 15);
        int kb = (lane < 16) ? 0 : 8;
        int k0 = ks * 32;
        _Float16* dst = A_pre + tid * 16;
        #pragma unroll
        for (int j = 0; j < 8; ++j)
            dst[j] = (_Float16)w_conv[o * KDIM + k0 + kb + j];
        #pragma unroll
        for (int j = 0; j < 8; ++j)
            dst[8 + j] = (_Float16)w_conv[o * KDIM + k0 + 16 + kb + j];
    } else if (tid < 880) {                // fused BN scale/shift, packed float2
        int o = tid - 800;
        float iv = gamma[o] * rsqrtf(var[o] + 1e-5f);
        invshift[2 * o]     = iv;
        invshift[2 * o + 1] = b_conv[o] * iv + beta[o] - mean[o] * iv;
    }
}

// ---------------- kernel 4: fused interp + GEMM + BN + residual + ReLU ------
// 320 threads = 10 waves; wave w -> (m_tile = w%5, n_tile = w/5); 32 sites/block.
__launch_bounds__(320)
__global__ void fused_prior_conv(const float* __restrict__ bev,
                                 const float* __restrict__ feats,
                                 const int* __restrict__ idxgrid,
                                 const _Float16* __restrict__ A_pre,
                                 const float* __restrict__ invshift,
                                 float* __restrict__ out) {
    __shared__ _Float16 Xt[NT][176];    // fp16 B panel [site][k], row 352B (16B aligned)
    __shared__ float    s_bev[NT][84];  // fp32 bev stage for residual (84: bank-depad)
    __shared__ int      s_cpt[NT * 8];  // per-site 8 corner point indices
    __shared__ float    s_cw[NT * 8];   // per-site 8 corner trilinear weights

    const int tid  = threadIdx.x;
    const int base = blockIdx.x * NT;   // first of 32 spatial sites for this block

    // ---- phase 0: trilinear corner setup (one thread per site) ----
    if (tid < NT) {
        int site = base + tid;
        int z = site & 15;
        int w = (site >> 4) % OUT_W;
        int h = site / (OUT_W * OUT_Z);

        float py = fminf(fmaxf((h + 0.5f) * (256.0f / 200.0f) - 0.5f, 0.0f), 255.0f);
        float px = fminf(fmaxf((w + 0.5f) * (256.0f / 200.0f) - 0.5f, 0.0f), 255.0f);
        float pz = fminf(fmaxf((z + 0.5f) * (20.0f / 16.0f)  - 0.5f, 0.0f), 19.0f);
        int y0 = (int)floorf(py); float fy = py - (float)y0; int y1 = min(y0 + 1, 255);
        int x0 = (int)floorf(px); float fx = px - (float)x0; int x1 = min(x0 + 1, 255);
        int z0 = (int)floorf(pz); float fz = pz - (float)z0; int z1 = min(z0 + 1, 19);

        #pragma unroll
        for (int j = 0; j < 8; ++j) {
            int xs = (j & 1) ? x1 : x0;  float wx = (j & 1) ? fx : 1.0f - fx;
            int ys = (j & 2) ? y1 : y0;  float wy = (j & 2) ? fy : 1.0f - fy;
            int zs = (j & 4) ? z1 : z0;  float wz = (j & 4) ? fz : 1.0f - fz;
            s_cpt[tid * 8 + j] = idxgrid[(xs * VOX_Y + ys) * VOX_Z + zs];
            s_cw[tid * 8 + j]  = wx * wy * wz;
        }
    }
    __syncthreads();

    // ---- phase 1: build fp16 B panel (k<80: bev rows, 128B coalesced; k>=80: prior) ----
    {
        const int n    = tid & 31;
        const int ksub = tid >> 5;                  // 0..9
        for (int pass = 0; pass < 16; ++pass) {
            int k = pass * 10 + ksub;               // 0..159, each exactly once
            float v;
            if (k < NCH) {
                v = bev[k * SPATIAL + base + n];    // full 128B line per row
                s_bev[n][k] = v;                    // stage fp32 residual in LDS
            } else {
                int c = k - NCH;
                v = 0.0f;
                #pragma unroll
                for (int j = 0; j < 8; ++j) {
                    int pt = s_cpt[n * 8 + j];
                    if (pt >= 0) v += s_cw[n * 8 + j] * feats[pt * NCH + c]; // L2-resident
                }
            }
            Xt[n][k] = (_Float16)v;
        }
    }
    __syncthreads();

    // ---- phase 2: wave (m_tile, n_tile); K=160 via 5 chained WMMAs ----
    const int wave   = tid >> 5;         // 0..9
    const int m_tile = wave % 5;
    const int n_tile = wave / 5;         // 0 or 1
    const int lane   = tid & 31;
    const int bn     = lane & 15;        // B-matrix column within the 16-wide tile
    const int n      = n_tile * 16 + bn; // site within block
    const int khalf  = (lane < 16) ? 0 : 8;

    v8f acc = {};
    for (int ks = 0; ks < 5; ++ks) {
        v16h a = *(const v16h*)(A_pre + ((m_tile * 5 + ks) * 32 + lane) * 16);
        const v8h* bp = (const v8h*)(&Xt[n][ks * 32 + khalf]);
        v8h b0 = bp[0];
        v8h b1 = bp[2];
        v16h b;
        #pragma unroll
        for (int i = 0; i < 8; ++i) { b[i] = b0[i]; b[i + 8] = b1[i]; }
        acc = __builtin_amdgcn_wmma_f32_16x16x32_f16(
            /*neg_a=*/false, a, /*neg_b=*/false, b,
            /*c_mod=*/(short)0, acc, /*reuse_a=*/false, /*reuse_b=*/false);
    }

    // ---- epilogue: BN + residual(LDS) + ReLU; D layout: VGPR r -> M = r + (lane<16?0:8) ----
    const int site  = base + n;
    const int mbase = m_tile * 16 + ((lane < 16) ? 0 : 8);
    const float2* ivs = (const float2*)invshift;
    #pragma unroll
    for (int r = 0; r < 8; ++r) {
        int o = mbase + r;
        float2 p = ivs[o];                           // (scale, shift) in one b64
        float y = acc[r] * p.x + p.y + s_bev[n][o];
        out[o * SPATIAL + site] = fmaxf(y, 0.0f);
    }
}

// ---------------- host-side launcher ----------------
extern "C" void kernel_launch(void* const* d_in, const int* in_sizes, int n_in,
                              void* d_out, int out_size, void* d_ws, size_t ws_size,
                              hipStream_t stream) {
    const float* bev     = (const float*)d_in[0];
    const int*   coords  = (const int*)  d_in[1];
    const float* feats   = (const float*)d_in[2];
    const float* w_conv  = (const float*)d_in[3];
    const float* b_conv  = (const float*)d_in[4];
    const float* gamma   = (const float*)d_in[5];
    const float* beta    = (const float*)d_in[6];
    const float* mean    = (const float*)d_in[7];
    const float* var     = (const float*)d_in[8];
    float* out = (float*)d_out;

    // workspace layout (aligned slices)
    char* ws = (char*)d_ws;
    int*      idxgrid  = (int*)ws;                          // 5,242,880 B
    _Float16* A_pre    = (_Float16*)(ws + 5242880);         //    25,600 B
    float*    invshift = (float*)   (ws + 5242880 + 25600); //       640 B

    init_idxgrid<<<(VOX_TOT + 255) / 256, 256, 0, stream>>>(idxgrid);
    scatter_points<<<(N_PTS + 255) / 256, 256, 0, stream>>>(coords, idxgrid);
    prep_weights<<<7, 128, 0, stream>>>(w_conv, b_conv, gamma, beta, mean, var,
                                        A_pre, invshift);
    fused_prior_conv<<<SPATIAL / NT, 320, 0, stream>>>(bev, feats, idxgrid,
                                                       A_pre, invshift, out);
}
